// LinkEncoder_75866302316988
// MI455X (gfx1250) — compile-verified, matrix operand order checked by reading, not run on hardware
//
#include <hip/hip_runtime.h>
#include <hip/hip_bf16.h>
#include <math.h>

// ---------------- problem constants (from reference) ----------------
#define NUM_NODES 30000
#define NUM_EDGES 600000
#define KLAT 20          // latest-K edges per node
#define HID 64
#define IN_CH 64
#define TIME_CH 64
#define ENC_IN 128       // TIME_CH + IN_CH
#define C1 256           // 4*HID channel-MLP hidden

typedef __attribute__((ext_vector_type(16))) _Float16 v16h;
typedef __attribute__((ext_vector_type(8)))  float    v8f;

// ---------------- WMMA fragment helpers (wave32, ISA 7.12.2) ----------------
// 16-bit A 16x32 fragment: lane -> m = lane&15; element pair p holds
// K = (p<4?0:16) + (lane>=16?8:0) + (p&3)*2  (and +1 for the odd half).
// Note: elements 0..7 and 8..15 are each 16 contiguous bytes per lane, so the
// compiler merges each fragment into two b128 loads.
__device__ __forceinline__ int frag_kidx(int lane, int p) {
  return ((p & 4) << 2) | ((lane >> 4) << 3) | ((p & 3) << 1);
}

// Loads a 16-half fragment from a row-major [rows][ld] array.
// For A: fixed = m (global row).  For B computing x@W^T: fixed = n, base = W
// (row-major [N][Ktot]) so that b[k][n] = W[n][k].
__device__ __forceinline__ v16h frag_load_rowmajor(const _Float16* base, int ld,
                                                   int fixed, int k0, int lane) {
  v16h f;
  const _Float16* row = base + (size_t)fixed * ld + k0;
#pragma unroll
  for (int p = 0; p < 8; ++p) {
    int kb = frag_kidx(lane, p);
    f[2 * p]     = row[kb];
    f[2 * p + 1] = row[kb + 1];
  }
  return f;
}

__device__ __forceinline__ v8f wmma_f16(v16h a, v16h b, v8f c) {
  // (neg_a, A, neg_b, B, c_mod, C, reuse_a, reuse_b)
  return __builtin_amdgcn_wmma_f32_16x16x32_f16(false, a, false, b, (short)0, c,
                                                false, false);
}

__device__ __forceinline__ float gelu_exact(float x) {
  return 0.5f * x * (1.0f + erff(x * 0.70710678118654752f));
}

// ---------------- small utility kernels ----------------
__global__ void zero_ints_kernel(int* p, int n) {
  int i = blockIdx.x * blockDim.x + threadIdx.x;
  if (i < n) p[i] = 0;
}

__global__ void convert_weights_kernel(const float* __restrict__ enc_w,
                                       const float* __restrict__ cl1_w,
                                       const float* __restrict__ cl2_w,
                                       _Float16* __restrict__ enc_w16,
                                       _Float16* __restrict__ cl1_w16,
                                       _Float16* __restrict__ cl2_w16) {
  int i = blockIdx.x * blockDim.x + threadIdx.x;
  if (i < HID * ENC_IN) enc_w16[i] = (_Float16)enc_w[i];
  if (i < C1 * HID)     cl1_w16[i] = (_Float16)cl1_w[i];
  if (i < HID * C1)     cl2_w16[i] = (_Float16)cl2_w[i];
}

// ---------------- 1) encoder: h = [cos(t*f) ; attr] @ enc_w^T + enc_b ----------------
// One block = 16 edges (grid divides NUM_EDGES exactly; no bounds guards).
// 4 waves each own a 16-wide slice of the 64 outputs. Results staged through
// LDS so the global writeback is coalesced float4 stores.
__global__ void encode_kernel(const float* __restrict__ edge_attr,
                              const int* __restrict__ edge_time,
                              const _Float16* __restrict__ enc_w16,
                              const float* __restrict__ enc_b,
                              float* __restrict__ h) {
  __shared__ _Float16 As[16 * ENC_IN];  // 16 edges x 128 features, f16
  __shared__ float Ds[16 * HID];        // 16 edges x 64 outputs, f32
  const int tile = blockIdx.x;
  const int tid = threadIdx.x;          // 0..127
  const int lane = tid & 31, wave = tid >> 5;
  const int col = tid;                  // feature index 0..127
  // freq[i] = 10^(-i/sqrt(64)) = 2^(-i/8 * log2(10))
  const float freq = (col < TIME_CH)
      ? exp2f(-(float)col * (0.125f * 3.321928094887362f)) : 0.0f;
#pragma unroll
  for (int r = 0; r < 16; ++r) {
    int e = tile * 16 + r;
    float v;
    if (col < TIME_CH) v = cosf((float)edge_time[e] * freq);
    else               v = edge_attr[(size_t)e * IN_CH + (col - TIME_CH)];
    As[r * ENC_IN + col] = (_Float16)v;
  }
  __syncthreads();

  const int n = wave * 16 + (lane & 15);       // output channel
  v8f c;
  const float bias = enc_b[n];
#pragma unroll
  for (int p = 0; p < 8; ++p) c[p] = bias;
#pragma unroll
  for (int ks = 0; ks < 4; ++ks) {             // K = 128 = 4 * 32
    v16h a = frag_load_rowmajor(As, ENC_IN, (lane & 15), ks * 32, lane);
    v16h b = frag_load_rowmajor(enc_w16, ENC_IN, n, ks * 32, lane);
    c = wmma_f16(a, b, c);
  }
  // D layout: row m = p + 8*(lane>>4), col n. Consecutive lanes hit
  // consecutive LDS columns -> conflict-free b32 stores.
#pragma unroll
  for (int p = 0; p < 8; ++p) {
    int m = p + ((lane >> 4) << 3);
    Ds[m * HID + n] = c[p];
  }
  __syncthreads();

  // coalesced writeback: 16*64 floats = 256 float4, 2 per thread
  float4* dst = (float4*)(h + (size_t)tile * 16 * HID);
  const float4* src = (const float4*)Ds;
#pragma unroll
  for (int i = 0; i < 2; ++i) dst[tid + i * 128] = src[tid + i * 128];
}

// ---------------- 2) group-by-destination machinery ----------------
__global__ void count_kernel(const int* __restrict__ col, int* __restrict__ deg,
                             int nE) {
  int e = blockIdx.x * blockDim.x + threadIdx.x;
  if (e < nE) atomicAdd(&deg[col[e]], 1);
}

// single-block chunked exclusive scan (30000 elements -> 30 chunks of 1024)
__global__ void scan_kernel(const int* __restrict__ deg, int* __restrict__ off,
                            int n) {
  __shared__ int s[1024];
  __shared__ int carry;
  if (threadIdx.x == 0) carry = 0;
  __syncthreads();
  for (int base = 0; base < n; base += 1024) {
    int i = base + threadIdx.x;
    int v = (i < n) ? deg[i] : 0;
    s[threadIdx.x] = v;
    __syncthreads();
    for (int d = 1; d < 1024; d <<= 1) {
      int t = (threadIdx.x >= d) ? s[threadIdx.x - d] : 0;
      __syncthreads();
      s[threadIdx.x] += t;
      __syncthreads();
    }
    int incl = s[threadIdx.x];
    int c0 = carry;
    int total = s[1023];
    __syncthreads();
    if (threadIdx.x == 0) carry = c0 + total;
    __syncthreads();
    if (i < n) off[i] = c0 + incl - v;   // exclusive prefix
  }
  if (threadIdx.x == 0) off[n] = carry;
}

__global__ void bucket_kernel(const int* __restrict__ col,
                              const int* __restrict__ off,
                              int* __restrict__ cur, int* __restrict__ bucket,
                              int nE) {
  int e = blockIdx.x * blockDim.x + threadIdx.x;
  if (e >= nE) return;
  int c = col[e];
  int p = atomicAdd(&cur[c], 1);
  bucket[off[c] + p] = e;
}

// per-node latest-K selection, ordered by (-time, edge_id) to match the
// reference's stable lexsort((-time, col)).
__global__ void topk_kernel(const int* __restrict__ off,
                            const int* __restrict__ bucket,
                            const int* __restrict__ etime,
                            int* __restrict__ slots, int nnodes) {
  int n = blockIdx.x * blockDim.x + threadIdx.x;
  if (n >= nnodes) return;
  int tt[KLAT];
  int ee[KLAT];
  int cnt = 0;
  int s = off[n], e = off[n + 1];
  for (int i = s; i < e; ++i) {
    int eid = bucket[i];
    int tv = etime[eid];
    bool take = (cnt < KLAT) || (tv > tt[KLAT - 1]) ||
                (tv == tt[KLAT - 1] && eid < ee[KLAT - 1]);
    if (take) {
      int pos = (cnt < KLAT - 1) ? cnt : (KLAT - 1);
      while (pos > 0 &&
             (tt[pos - 1] < tv || (tt[pos - 1] == tv && ee[pos - 1] > eid))) {
        tt[pos] = tt[pos - 1];
        ee[pos] = ee[pos - 1];
        --pos;
      }
      tt[pos] = tv;
      ee[pos] = eid;
      if (cnt < KLAT) ++cnt;
    }
  }
  for (int r = 0; r < KLAT; ++r)
    slots[n * KLAT + r] = (r < cnt) ? ee[r] : -1;
}

// ---------------- 3) fused per-node MLP-mixer + head ----------------
// One block (128 threads / 4 waves) per node. Channel MLP runs on WMMA with
// the 20-row activation padded to 32 rows (M-tiles of 16). All WMMA D stores
// are unconditional (padded scratch); residuals are folded into later phases.
__global__ void mixer_kernel(const float* __restrict__ h,
                             const int* __restrict__ slots,
                             const _Float16* __restrict__ cl1_w16,
                             const float* __restrict__ cl1_b,
                             const _Float16* __restrict__ cl2_w16,
                             const float* __restrict__ cl2_b,
                             const float* __restrict__ tl1_w,
                             const float* __restrict__ tl1_b,
                             const float* __restrict__ tl2_w,
                             const float* __restrict__ tl2_b,
                             const float* __restrict__ tok_g,
                             const float* __restrict__ tok_b,
                             const float* __restrict__ ch_g,
                             const float* __restrict__ ch_b,
                             const float* __restrict__ hd_g,
                             const float* __restrict__ hd_b,
                             const float* __restrict__ head_w,
                             const float* __restrict__ head_b,
                             float* __restrict__ out) {
  __shared__ float xs[KLAT * HID];     // gathered x (residual 1)
  __shared__ float ht[KLAT * HID];     // h_tok (residual 2)
  __shared__ _Float16 a16[32 * HID];   // LN(h_tok), padded rows, A for cl1
  __shared__ _Float16 u16[32 * C1];    // gelu(cl1), A for cl2
  __shared__ float vsh[32 * HID];      // raw cl2 output (padded rows)
  __shared__ float mrow[KLAT], rrow[KLAT];
  __shared__ float meanv[HID];
  __shared__ int sslot[KLAT];

  const int node = blockIdx.x;
  const int tid = threadIdx.x, lane = tid & 31, wave = tid >> 5;

  if (tid < KLAT) sslot[tid] = slots[node * KLAT + tid];
  __syncthreads();

  // gather x[20][64] from h
  for (int i = tid; i < KLAT * HID; i += 128) {
    int r = i >> 6, c = i & 63;
    int e = sslot[r];
    xs[i] = (e >= 0) ? h[(size_t)e * HID + c] : 0.0f;
  }
  __syncthreads();

  // ---- token LN stats ----
  if (tid < KLAT) {
    float s = 0.f, s2 = 0.f;
    for (int c = 0; c < HID; ++c) { float v = xs[tid * HID + c]; s += v; s2 += v * v; }
    float m = s * (1.0f / HID);
    float var = s2 * (1.0f / HID) - m * m;
    mrow[tid] = m;
    rrow[tid] = rsqrtf(var + 1e-5f);
  }
  __syncthreads();

  // ---- token MLP (20 -> 10 -> 20) per channel, VALU ----
  if (tid < HID) {
    float zn[KLAT];
    const float g = tok_g[tid], bb = tok_b[tid];
#pragma unroll
    for (int k = 0; k < KLAT; ++k)
      zn[k] = (xs[k * HID + tid] - mrow[k]) * rrow[k] * g + bb;
    float t1[KLAT / 2];
#pragma unroll
    for (int j = 0; j < KLAT / 2; ++j) {
      float acc = tl1_b[j];
#pragma unroll
      for (int k = 0; k < KLAT; ++k) acc += zn[k] * tl1_w[j * KLAT + k];
      t1[j] = gelu_exact(acc);
    }
#pragma unroll
    for (int k = 0; k < KLAT; ++k) {
      float acc = tl2_b[k];
#pragma unroll
      for (int j = 0; j < KLAT / 2; ++j) acc += t1[j] * tl2_w[k * (KLAT / 2) + j];
      ht[k * HID + tid] = acc + xs[k * HID + tid];
    }
  }
  __syncthreads();

  // ---- channel LN stats ----
  if (tid < KLAT) {
    float s = 0.f, s2 = 0.f;
    for (int c = 0; c < HID; ++c) { float v = ht[tid * HID + c]; s += v; s2 += v * v; }
    float m = s * (1.0f / HID);
    float var = s2 * (1.0f / HID) - m * m;
    mrow[tid] = m;
    rrow[tid] = rsqrtf(var + 1e-5f);
  }
  __syncthreads();
  // fill padded f16 A-matrix (rows 20..31 zero; their outputs are discarded)
  for (int i = tid; i < 32 * HID; i += 128) {
    int r = i >> 6, c = i & 63;
    float v = 0.0f;
    if (r < KLAT) v = (ht[r * HID + c] - mrow[r]) * rrow[r] * ch_g[c] + ch_b[c];
    a16[i] = (_Float16)v;
  }
  __syncthreads();

  // ---- cl1: [32,64] @ [64,256] -> gelu -> u16 ; 32 tiles, 8 per wave ----
  for (int t = wave * 8; t < wave * 8 + 8; ++t) {
    int mt = t & 1, nt = t >> 1;
    int nn = nt * 16 + (lane & 15);
    v8f c;
    float bb = cl1_b[nn];
#pragma unroll
    for (int p = 0; p < 8; ++p) c[p] = bb;
#pragma unroll
    for (int ks = 0; ks < 2; ++ks) {
      v16h a = frag_load_rowmajor(a16, HID, mt * 16 + (lane & 15), ks * 32, lane);
      v16h b = frag_load_rowmajor(cl1_w16, HID, nn, ks * 32, lane);
      c = wmma_f16(a, b, c);
    }
#pragma unroll
    for (int p = 0; p < 8; ++p) {
      int m = mt * 16 + p + ((lane >> 4) << 3);
      u16[m * C1 + nn] = (_Float16)gelu_exact(c[p]);
    }
  }
  __syncthreads();

  // ---- cl2: [32,256] @ [256,64] -> vsh (raw) ; 8 tiles, 2 per wave ----
  for (int t = wave * 2; t < wave * 2 + 2; ++t) {
    int mt = t & 1, nt = t >> 1;
    int nn = nt * 16 + (lane & 15);
    v8f c;
    float bb = cl2_b[nn];
#pragma unroll
    for (int p = 0; p < 8; ++p) c[p] = bb;
#pragma unroll
    for (int ks = 0; ks < 8; ++ks) {
      v16h a = frag_load_rowmajor(u16, C1, mt * 16 + (lane & 15), ks * 32, lane);
      v16h b = frag_load_rowmajor(cl2_w16, C1, nn, ks * 32, lane);
      c = wmma_f16(a, b, c);
    }
#pragma unroll
    for (int p = 0; p < 8; ++p) {
      int m = mt * 16 + p + ((lane >> 4) << 3);
      vsh[m * HID + nn] = c[p];          // unconditional; padded rows unused
    }
  }
  __syncthreads();

  // ---- head LN stats on (cl2 + h_tok residual) ----
  if (tid < KLAT) {
    float s = 0.f, s2 = 0.f;
    for (int c = 0; c < HID; ++c) {
      float v = vsh[tid * HID + c] + ht[tid * HID + c];
      s += v; s2 += v * v;
    }
    float m = s * (1.0f / HID);
    float var = s2 * (1.0f / HID) - m * m;
    mrow[tid] = m;
    rrow[tid] = rsqrtf(var + 1e-5f);
  }
  __syncthreads();
  // mean over K of LN rows
  if (tid < HID) {
    const float g = hd_g[tid], bb = hd_b[tid];
    float acc = 0.f;
#pragma unroll
    for (int k = 0; k < KLAT; ++k) {
      float v = vsh[k * HID + tid] + ht[k * HID + tid];
      acc += (v - mrow[k]) * rrow[k] * g + bb;
    }
    meanv[tid] = acc * (1.0f / KLAT);
  }
  __syncthreads();
  // head: out = mean @ head_w^T + head_b
  if (tid < HID) {
    float acc = head_b[tid];
#pragma unroll
    for (int c = 0; c < HID; ++c) acc += meanv[c] * head_w[tid * HID + c];
    out[(size_t)node * HID + tid] = acc;
  }
}

// ---------------- launch ----------------
extern "C" void kernel_launch(void* const* d_in, const int* in_sizes, int n_in,
                              void* d_out, int out_size, void* d_ws,
                              size_t ws_size, hipStream_t stream) {
  (void)in_sizes; (void)n_in; (void)out_size; (void)ws_size;
  // setup_inputs() dict order with the nested params dict flattened in
  // insertion order:
  const float* edge_attr = (const float*)d_in[0];
  const float* enc_w  = (const float*)d_in[1];
  const float* enc_b  = (const float*)d_in[2];
  const float* tl1_w  = (const float*)d_in[3];
  const float* tl1_b  = (const float*)d_in[4];
  const float* tl2_w  = (const float*)d_in[5];
  const float* tl2_b  = (const float*)d_in[6];
  const float* cl1_w  = (const float*)d_in[7];
  const float* cl1_b  = (const float*)d_in[8];
  const float* cl2_w  = (const float*)d_in[9];
  const float* cl2_b  = (const float*)d_in[10];
  const float* head_w = (const float*)d_in[11];
  const float* head_b = (const float*)d_in[12];
  const float* tok_g  = (const float*)d_in[13];
  const float* tok_b  = (const float*)d_in[14];
  const float* ch_g   = (const float*)d_in[15];
  const float* ch_b   = (const float*)d_in[16];
  const float* hd_g   = (const float*)d_in[17];
  const float* hd_b   = (const float*)d_in[18];
  const int* edge_time  = (const int*)d_in[19];
  const int* edge_index = (const int*)d_in[20];
  float* out = (float*)d_out;
  const int* col = edge_index + NUM_EDGES;  // edge_index[1]

  // workspace layout
  char* w = (char*)d_ws;
  auto alignup = [](size_t x) { return (x + 255) & ~(size_t)255; };
  float* h_buf = (float*)w;        w += alignup((size_t)NUM_EDGES * HID * 4);
  int* deg = (int*)w;              w += alignup((size_t)NUM_NODES * 4);
  int* off = (int*)w;              w += alignup((size_t)(NUM_NODES + 1) * 4);
  int* cur = (int*)w;              w += alignup((size_t)NUM_NODES * 4);
  int* bucket = (int*)w;           w += alignup((size_t)NUM_EDGES * 4);
  int* slots = (int*)w;            w += alignup((size_t)NUM_NODES * KLAT * 4);
  _Float16* enc_w16 = (_Float16*)w; w += alignup((size_t)HID * ENC_IN * 2);
  _Float16* cl1_w16 = (_Float16*)w; w += alignup((size_t)C1 * HID * 2);
  _Float16* cl2_w16 = (_Float16*)w; w += alignup((size_t)HID * C1 * 2);

  convert_weights_kernel<<<(C1 * HID + 255) / 256, 256, 0, stream>>>(
      enc_w, cl1_w, cl2_w, enc_w16, cl1_w16, cl2_w16);
  zero_ints_kernel<<<(NUM_NODES + 255) / 256, 256, 0, stream>>>(deg, NUM_NODES);
  zero_ints_kernel<<<(NUM_NODES + 255) / 256, 256, 0, stream>>>(cur, NUM_NODES);

  encode_kernel<<<NUM_EDGES / 16, 128, 0, stream>>>(
      edge_attr, edge_time, enc_w16, enc_b, h_buf);

  count_kernel<<<(NUM_EDGES + 255) / 256, 256, 0, stream>>>(col, deg, NUM_EDGES);
  scan_kernel<<<1, 1024, 0, stream>>>(deg, off, NUM_NODES);
  bucket_kernel<<<(NUM_EDGES + 255) / 256, 256, 0, stream>>>(col, off, cur,
                                                             bucket, NUM_EDGES);
  topk_kernel<<<(NUM_NODES + 127) / 128, 128, 0, stream>>>(off, bucket,
                                                           edge_time, slots,
                                                           NUM_NODES);

  mixer_kernel<<<NUM_NODES, 128, 0, stream>>>(
      h_buf, slots, cl1_w16, cl1_b, cl2_w16, cl2_b, tl1_w, tl1_b, tl2_w, tl2_b,
      tok_g, tok_b, ch_g, ch_b, hd_g, hd_b, head_w, head_b, out);
}
